// CausalSelfAttention_14800457302737
// MI455X (gfx1250) — matrix operand order repaired and checked by reading, not compiled, not run on hardware
//
#include <hip/hip_runtime.h>
#include <hip/hip_bf16.h>
#include <math.h>
#include <stdint.h>

typedef __bf16 bf16;
typedef __attribute__((ext_vector_type(8)))  __bf16 v8bf;
typedef __attribute__((ext_vector_type(16))) __bf16 v16bf;
typedef __attribute__((ext_vector_type(8)))  float  v8f;

#define S_LEN  2048
#define BATCH  2
#define DMODEL 2048
#define NH     16
#define DH     128

static __device__ __forceinline__ v8f zero8() {
  v8f z = {0.f, 0.f, 0.f, 0.f, 0.f, 0.f, 0.f, 0.f};
  return z;
}

// A-fragment: 16x32 bf16, row-major source, lane holds row M=lane%16,
// K split as {half*8..+7, 16+half*8..+7} per ISA table.
static __device__ __forceinline__ v16bf load_frag_a(const bf16* __restrict__ src,
                                                    int ld, int row0, int k0) {
  const int lane = threadIdx.x & 31;
  const int m = lane & 15, h = lane >> 4;
  const bf16* p = src + (size_t)(row0 + m) * ld + k0 + h * 8;
  v8bf lo = *(const v8bf*)p;
  v8bf hi = *(const v8bf*)(p + 16);
  v16bf r;
#pragma unroll
  for (int i = 0; i < 8; ++i) { r[i] = lo[i]; r[i + 8] = hi[i]; }
  return r;
}

// B-fragment: 32x16 bf16; source is row-major [N][K] (weight-style), lane holds
// column N=lane%16, K split as {half*8..+7, 16+half*8..+7} per ISA table.
static __device__ __forceinline__ v16bf load_frag_b(const bf16* __restrict__ src,
                                                    int ld, int n0, int k0) {
  const int lane = threadIdx.x & 31;
  const int n = lane & 15, h = lane >> 4;
  const bf16* p = src + (size_t)(n0 + n) * ld + k0 + h * 8;
  v8bf lo = *(const v8bf*)p;
  v8bf hi = *(const v8bf*)(p + 16);
  v16bf r;
#pragma unroll
  for (int i = 0; i < 8; ++i) { r[i] = lo[i]; r[i + 8] = hi[i]; }
  return r;
}

static __device__ __forceinline__ v8f wmma_bf16(v16bf a, v16bf b, v8f c) {
  return __builtin_amdgcn_wmma_f32_16x16x32_bf16(false, a, false, b, (short)0, c,
                                                 false, false);
}

// Async copy: 16 bytes global -> LDS, tracked by ASYNCcnt (CDNA5 path).
static __device__ __forceinline__ void async_copy_b128(unsigned lds_off,
                                                       const bf16* gaddr) {
  asm volatile("global_load_async_to_lds_b128 %0, %1, off"
               :: "v"(lds_off), "v"(gaddr) : "memory");
}

// ---------------------------------------------------------------- convert
__global__ __launch_bounds__(256) void f32_to_bf16_k(const float* __restrict__ in,
                                                     bf16* __restrict__ out, int n) {
  int i = blockIdx.x * blockDim.x + threadIdx.x;
  if (i < n) out[i] = (bf16)in[i];
}

// ---------------------------------------------------------------- GEMM
// C[M,N] = A[M,K] * Bw[N,K]^T  (bf16 in, f32 out). Block = 8 waves, tile 64x256,
// double-buffered LDS staging via async global->LDS copies.
__global__ __launch_bounds__(256) void gemm_bf16_nt(const bf16* __restrict__ A,
                                                    const bf16* __restrict__ Bw,
                                                    float* __restrict__ C,
                                                    int K, int N) {
  __shared__ __align__(16) bf16 sA[2][64][32];    // 2 x 4KB
  __shared__ __align__(16) bf16 sB[2][256][32];   // 2 x 16KB

  const int tid = threadIdx.x;
  const int wave = tid >> 5;
  const int lane = tid & 31;
  const int wm = wave >> 2;                       // 0..1
  const int wn = wave & 3;                        // 0..3
  const int rowblk = blockIdx.y * 64;
  const int colblk = blockIdx.x * 256;
  const int row0 = wm * 32;                       // within LDS tile
  const int col0 = wn * 64;

  // per-thread copy assignment: A tile 64x32 (4 threads/row, 16B each),
  // B tile 256x32 (1 row/thread, 4x16B).
  const int ar = tid >> 2;
  const int ac = (tid & 3) * 8;
  const unsigned ldsA[2] = {(unsigned)(uintptr_t)&sA[0][ar][ac],
                            (unsigned)(uintptr_t)&sA[1][ar][ac]};
  const unsigned ldsB[2] = {(unsigned)(uintptr_t)&sB[0][tid][0],
                            (unsigned)(uintptr_t)&sB[1][tid][0]};
  const bf16* gA = A + (size_t)(rowblk + ar) * K + ac;
  const bf16* gB = Bw + (size_t)(colblk + tid) * K;

  v8f acc[2][4];
#pragma unroll
  for (int i = 0; i < 2; ++i)
#pragma unroll
    for (int j = 0; j < 4; ++j) acc[i][j] = zero8();

  const int nk = K >> 5;

  // prime stage 0
  {
    async_copy_b128(ldsA[0], gA);
    async_copy_b128(ldsB[0] +  0, gB);
    async_copy_b128(ldsB[0] + 16, gB + 8);
    async_copy_b128(ldsB[0] + 32, gB + 16);
    async_copy_b128(ldsB[0] + 48, gB + 24);
  }

  for (int kb = 0; kb < nk; ++kb) {
    const int buf = kb & 1;
    const bool more = (kb + 1) < nk;
    if (more) {
      const int k = (kb + 1) << 5;
      const int nb = buf ^ 1;
      async_copy_b128(ldsA[nb], gA + k);
      async_copy_b128(ldsB[nb] +  0, gB + k);
      async_copy_b128(ldsB[nb] + 16, gB + k + 8);
      async_copy_b128(ldsB[nb] + 32, gB + k + 16);
      async_copy_b128(ldsB[nb] + 48, gB + k + 24);
      asm volatile("s_wait_asynccnt 0x5" ::: "memory");  // stage kb done (mine)
    } else {
      asm volatile("s_wait_asynccnt 0x0" ::: "memory");
    }
    __syncthreads();                                     // stage kb done (all waves)

    const bf16* tA = &sA[buf][0][0];
    const bf16* tB = &sB[buf][0][0];
    v16bf a0 = load_frag_a(tA, 32, row0, 0);
    v16bf a1 = load_frag_a(tA, 32, row0 + 16, 0);
    v16bf b[4];
#pragma unroll
    for (int j = 0; j < 4; ++j) b[j] = load_frag_b(tB, 32, col0 + j * 16, 0);
#pragma unroll
    for (int j = 0; j < 4; ++j) {
      acc[0][j] = wmma_bf16(a0, b[j], acc[0][j]);
      acc[1][j] = wmma_bf16(a1, b[j], acc[1][j]);
    }
    __syncthreads();                                     // done reading buf before reuse
  }

  const int n = lane & 15, h = lane >> 4;
#pragma unroll
  for (int i = 0; i < 2; ++i)
#pragma unroll
    for (int j = 0; j < 4; ++j)
#pragma unroll
      for (int r = 0; r < 8; ++r)
        C[(size_t)(rowblk + row0 + i * 16 + r + 8 * h) * N +
          colblk + col0 + j * 16 + n] = acc[i][j][r];
}

// ---------------------------------------------------------------- RoPE + scatter
__global__ __launch_bounds__(256) void rope_scatter(const float* __restrict__ qkv,
                                                    bf16* __restrict__ q_bh,
                                                    bf16* __restrict__ k_bh,
                                                    bf16* __restrict__ vT_bh) {
  int idx = blockIdx.x * blockDim.x + threadIdx.x;       // B*NH*S*(DH/2) = 2^22
  int i = idx & 63;                                      // rope pair
  int s = (idx >> 6) & (S_LEN - 1);
  int h = (idx >> 17) & (NH - 1);
  int b = idx >> 21;
  if (b >= BATCH) return;

  const size_t t = (size_t)s * BATCH + b;
  const float* row = qkv + t * (3 * DMODEL);
  const int col = h * DH + 2 * i;
  float q0 = row[col],              q1 = row[col + 1];
  float k0 = row[DMODEL + col],     k1 = row[DMODEL + col + 1];
  float v0 = row[2 * DMODEL + col], v1 = row[2 * DMODEL + col + 1];

  // inv_freq = 10000^(-2i/128) = exp(-ln(10000) * 2i/128)
  float freq = __expf(-9.210340371976184f * (float)(2 * i) / (float)DH);
  float ang = (float)s * freq;
  float c = __cosf(ang), sn = __sinf(ang);
  const float scale = 0.08838834764831845f;              // 1/sqrt(128) folded into Q

  float qe0 = (q0 * c - q1 * sn) * scale;
  float qe1 = (q0 * sn + q1 * c) * scale;
  float ke0 = k0 * c - k1 * sn;
  float ke1 = k0 * sn + k1 * c;

  size_t bh = (size_t)b * NH + h;
  size_t o  = (bh * S_LEN + s) * DH + 2 * i;
  q_bh[o] = (bf16)qe0; q_bh[o + 1] = (bf16)qe1;
  k_bh[o] = (bf16)ke0; k_bh[o + 1] = (bf16)ke1;
  size_t ov = (bh * DH + 2 * i) * S_LEN + s;             // V stored transposed [d][s]
  vT_bh[ov] = (bf16)v0; vT_bh[ov + S_LEN] = (bf16)v1;
}

// ---------------------------------------------------------------- flash attention
// Block = 4 waves; each wave owns a 16-row query tile, streams 32-key blocks.
// Only wave-private sync (waves have different causal trip counts).
__global__ __launch_bounds__(128) void flash_attn(const bf16* __restrict__ q_bh,
                                                  const bf16* __restrict__ k_bh,
                                                  const bf16* __restrict__ vT_bh,
                                                  const unsigned char* __restrict__ seq_mask,
                                                  bf16* __restrict__ attn_out) {
  __shared__ __align__(16) float s_tile[4][16][32];
  __shared__ __align__(16) bf16  p_tile[4][16][32];
  __shared__ float s_m[4][16], s_l[4][16], s_corr[4][16];

  const int wave = threadIdx.x >> 5;
  const int lane = threadIdx.x & 31;
  const int n = lane & 15, half = lane >> 4;
  const int bh = blockIdx.y;
  const int b = bh >> 4;
  const int h = bh & 15;
  const int q0 = blockIdx.x * 64 + wave * 16;

  const bf16* qbase = q_bh + (size_t)bh * S_LEN * DH;
  const bf16* kbase = k_bh + (size_t)bh * S_LEN * DH;
  const bf16* vbase = vT_bh + (size_t)bh * DH * S_LEN;

  v16bf qf[4];
#pragma unroll
  for (int st = 0; st < 4; ++st) qf[st] = load_frag_a(qbase, DH, q0, st * 32);

  v8f oacc[8];
#pragma unroll
  for (int j = 0; j < 8; ++j) oacc[j] = zero8();

  if (lane < 16) { s_m[wave][lane] = -INFINITY; s_l[wave][lane] = 0.f; }
  asm volatile("s_wait_dscnt 0x0" ::: "memory");
  __builtin_amdgcn_wave_barrier();

  const int kb_max = (q0 + 15) >> 5;                     // causal upper bound
  for (int kb = 0; kb <= kb_max; ++kb) {
    const int k0 = kb * 32;
    v8f sacc0 = zero8(), sacc1 = zero8();
#pragma unroll
    for (int st = 0; st < 4; ++st) {
      v16bf b0 = load_frag_b(kbase, DH, k0, st * 32);
      v16bf b1 = load_frag_b(kbase, DH, k0 + 16, st * 32);
      sacc0 = wmma_bf16(qf[st], b0, sacc0);
      sacc1 = wmma_bf16(qf[st], b1, sacc1);
    }
    // mask + spill scores to LDS
#pragma unroll
    for (int r = 0; r < 8; ++r) {
      const int m = r + 8 * half;
      const int qrow = q0 + m;
      const int kc0 = k0 + n, kc1 = k0 + 16 + n;
      float v0s = sacc0[r], v1s = sacc1[r];
      if (kc0 > qrow || !seq_mask[b * S_LEN + kc0]) v0s = -INFINITY;
      if (kc1 > qrow || !seq_mask[b * S_LEN + kc1]) v1s = -INFINITY;
      s_tile[wave][m][n] = v0s;
      s_tile[wave][m][16 + n] = v1s;
    }
    asm volatile("s_wait_dscnt 0x0" ::: "memory");
    __builtin_amdgcn_wave_barrier();

    if (lane < 16) {                                     // online softmax, 1 lane/row
      const int m = lane;
      float mo = s_m[wave][m];
      float mx = mo;
#pragma unroll
      for (int kk = 0; kk < 32; ++kk) mx = fmaxf(mx, s_tile[wave][m][kk]);
      float corr = (mx == -INFINITY) ? 1.f : __expf(mo - mx);
      float lsum = s_l[wave][m] * corr;
#pragma unroll
      for (int kk = 0; kk < 32; ++kk) {
        float sv = s_tile[wave][m][kk];
        float e = (sv == -INFINITY) ? 0.f : __expf(sv - mx);
        p_tile[wave][m][kk] = (bf16)e;
        lsum += e;
      }
      s_m[wave][m] = mx; s_l[wave][m] = lsum; s_corr[wave][m] = corr;
    }
    asm volatile("s_wait_dscnt 0x0" ::: "memory");
    __builtin_amdgcn_wave_barrier();

#pragma unroll
    for (int r = 0; r < 8; ++r) {                        // rescale running O
      float f = s_corr[wave][r + 8 * half];
#pragma unroll
      for (int j = 0; j < 8; ++j) oacc[j][r] *= f;
    }
    v16bf pf = load_frag_a(&p_tile[wave][0][0], 32, 0, 0);
#pragma unroll
    for (int j = 0; j < 8; ++j) {                        // O += P x V
      v16bf bv = load_frag_b(vbase, S_LEN, j * 16, k0);
      oacc[j] = wmma_bf16(pf, bv, oacc[j]);
    }
  }

  asm volatile("s_wait_dscnt 0x0" ::: "memory");
  __builtin_amdgcn_wave_barrier();
#pragma unroll
  for (int r = 0; r < 8; ++r) {
    const int m = r + 8 * half;
    float l = s_l[wave][m];
    float inv = (l > 0.f) ? 1.f / l : 0.f;
    bf16* dst = attn_out + ((size_t)(q0 + m) * BATCH + b) * DMODEL + h * DH;
#pragma unroll
    for (int j = 0; j < 8; ++j) dst[j * 16 + n] = (bf16)(oacc[j][r] * inv);
  }
}

// ---------------------------------------------------------------- launch
extern "C" void kernel_launch(void* const* d_in, const int* in_sizes, int n_in,
                              void* d_out, int out_size, void* d_ws, size_t ws_size,
                              hipStream_t stream) {
  (void)in_sizes; (void)n_in; (void)out_size; (void)ws_size;
  const float* hidden = (const float*)d_in[0];
  const unsigned char* seq_mask = (const unsigned char*)d_in[1];
  const float* qkv_w = (const float*)d_in[2];
  const float* o_w   = (const float*)d_in[3];
  float* out = (float*)d_out;

  char* ws = (char*)d_ws;                                // 256B-aligned carve-out
  bf16*  hid_bf  = (bf16*)(ws + 0);                      // 4096*2048   bf16
  bf16*  qkvw_bf = (bf16*)(ws + 16777216);               // 6144*2048   bf16
  bf16*  ow_bf   = (bf16*)(ws + 41943040);               // 2048*2048   bf16
  float* qkv_f   = (float*)(ws + 50331648);              // 4096*6144   f32
  bf16*  q_bh    = (bf16*)(ws + 150994944);              // [B][H][S][DH]
  bf16*  k_bh    = (bf16*)(ws + 167772160);              // [B][H][S][DH]
  bf16*  vT_bh   = (bf16*)(ws + 184549376);              // [B][H][DH][S]
  bf16*  attn_bf = (bf16*)(ws + 201326592);              // [S][B][D]

  const int n_hid = 4096 * 2048, n_qw = 6144 * 2048, n_ow = 2048 * 2048;
  f32_to_bf16_k<<<(n_hid + 255) / 256, 256, 0, stream>>>(hidden, hid_bf, n_hid);
  f32_to_bf16_k<<<(n_qw + 255) / 256, 256, 0, stream>>>(qkv_w, qkvw_bf, n_qw);
  f32_to_bf16_k<<<(n_ow + 255) / 256, 256, 0, stream>>>(o_w, ow_bf, n_ow);

  // QKV projection: [4096,2048] x [6144,2048]^T
  gemm_bf16_nt<<<dim3(6144 / 256, 4096 / 64), 256, 0, stream>>>(hid_bf, qkvw_bf,
                                                                qkv_f, 2048, 6144);
  // RoPE + layout scatter
  rope_scatter<<<(BATCH * NH * S_LEN * (DH / 2)) / 256, 256, 0, stream>>>(
      qkv_f, q_bh, k_bh, vT_bh);
  // Flash attention: grid = (S/64 q-tiles, B*H)
  flash_attn<<<dim3(S_LEN / 64, BATCH * NH), 128, 0, stream>>>(q_bh, k_bh, vT_bh,
                                                               seq_mask, attn_bf);
  // Output projection: [4096,2048] x [2048,2048]^T -> fp32 out
  gemm_bf16_nt<<<dim3(2048 / 256, 4096 / 64), 256, 0, stream>>>(attn_bf, ow_bf,
                                                                out, 2048, 2048);
}
